// ReservoirController_1520418423029
// MI455X (gfx1250) — matrix operand order repaired and checked by reading, not compiled, not run on hardware
//
#include <hip/hip_runtime.h>
#include <hip/hip_bf16.h>
#include <math.h>

// Problem constants (from reference): B=128 batch, T=512 steps, U=2048 units, O=2 outputs.
#define BATCH 128
#define TLEN  512
#define UNITS 2048
#define LEAK  0.3f
#define KEEP  0.7f

// GEMM tiling for the recurrent step
#define KC    64          // K-chunk staged in LDS per iteration
#define AROWP 72          // padded LDS row length (elements): 144B = 36 banks -> conflict-free frag reads
#define BM    64          // workgroup M tile (2 waves)
#define BN    128         // workgroup N tile (4 waves)

typedef __attribute__((ext_vector_type(16))) __bf16 v16bf;
typedef __attribute__((ext_vector_type(8)))  __bf16 v8bf;
typedef __attribute__((ext_vector_type(8)))  float  v8f;

// ---------------------------------------------------------------------------
// One-time conversion: W_res f32 -> bf16 (stays L2-resident, 8MB << 192MB L2)
// ---------------------------------------------------------------------------
__global__ void esn_convert_w(const float* __restrict__ w, __bf16* __restrict__ wb, int n) {
    int i = blockIdx.x * blockDim.x + threadIdx.x;
    if (i < n) wb[i] = (__bf16)w[i];
}

__global__ void esn_init_h(float* __restrict__ hf, __bf16* __restrict__ hb, int n) {
    int i = blockIdx.x * blockDim.x + threadIdx.x;
    if (i < n) { hf[i] = 0.0f; hb[i] = (__bf16)0.0f; }
}

// ---------------------------------------------------------------------------
// CDNA5 async global->LDS copy (VGLOBAL async encoding, tracked by ASYNCcnt):
//   copies 16 bytes from a global address straight into an LDS address, no VGPR data staging.
// Generic LDS pointer low 32 bits == LDS byte address (flat-aperture truncation).
// ---------------------------------------------------------------------------
__device__ __forceinline__ void async_copy_b128(const __bf16* g, __bf16* l) {
    unsigned            lds = (unsigned)(size_t)l;
    unsigned long long  ga  = (unsigned long long)(size_t)g;
    asm volatile("global_load_async_to_lds_b128 %0, %1, off" :: "v"(lds), "v"(ga) : "memory");
}
__device__ __forceinline__ void wait_asynccnt0() {
    asm volatile("s_wait_asynccnt 0" ::: "memory");
}

// ---------------------------------------------------------------------------
// A-fragment (16-bit A 16x32 per-lane layout, ISA 7.12.2):
//   lane<16 : elems 0..7 = A[m][k0..k0+7],    elems 8..15 = A[m][k0+16..k0+23]
//   lane>=16: elems 0..7 = A[m][k0+8..k0+15], elems 8..15 = A[m][k0+24..k0+31]
// caller passes base = &A[m][k0 + 8*(lane>>4)] ; works for global or LDS pointers.
// ---------------------------------------------------------------------------
__device__ __forceinline__ v16bf load_frag_a(const __bf16* base) {
    v8bf lo = *(const v8bf*)base;
    v8bf hi = *(const v8bf*)(base + 16);
    return __builtin_shufflevector(lo, hi, 0,1,2,3,4,5,6,7,8,9,10,11,12,13,14,15);
}

// B-fragment (K x N = 32x16): lane = N column; same K order as A fragment:
// elems 0..7 = K=k0+8*(lane>>4)+0..7, elems 8..15 = K=k0+16+8*(lane>>4)+0..7.
__device__ __forceinline__ v16bf load_frag_b(const __bf16* base) {
    v8bf lo = *(const v8bf*)base;
    v8bf hi = *(const v8bf*)(base + 16);
    return __builtin_shufflevector(lo, hi, 0,1,2,3,4,5,6,7,8,9,10,11,12,13,14,15);
}

// ---------------------------------------------------------------------------
// One recurrence step: pre = h@W_res^T + x[:,t]*W_in^T ; h' = 0.7h + 0.3 tanh(pre)
//
// LDS-staged NT GEMM, M=128 N=2048 K=2048:
//   workgroup = 8 waves (2 M-waves x 4 N-waves), tile 64(M) x 128(N)
//   per wave: 32x32 output via 2x2 v_wmma_f32_16x16x32_bf16 accumulators
//   K staged in KC=64 chunks, double-buffered LDS filled by ASYNC global->LDS
//   grid = (2048/128, 128/64) = (16,2) = 32 WGs = 256 waves
// L2 traffic/step: A 8MB + B 16MB (vs 64MB unstaged); one barrier per chunk.
// ---------------------------------------------------------------------------
__global__ void __launch_bounds__(256) esn_step(
        const __bf16* __restrict__ hb_src,   // h in bf16, [BATCH][UNITS]
        const __bf16* __restrict__ Wb,       // W_res in bf16, [UNITS][UNITS] (row n, K contiguous)
        const float*  __restrict__ hf_src,   // h master f32, [BATCH][UNITS]
        const float*  __restrict__ x,        // [BATCH][TLEN]
        const float*  __restrict__ W_in,     // [UNITS]
        int t,
        float*  __restrict__ hf_dst,
        __bf16* __restrict__ hb_dst) {
    __shared__ __align__(16) __bf16 As[2][BM * AROWP];   // 2 x 9216B
    __shared__ __align__(16) __bf16 Bs[2][BN * AROWP];   // 2 x 18432B

    const int tid  = threadIdx.x;
    const int lane = tid & 31;
    const int wave = tid >> 5;
    const int wm   = wave >> 2;        // 0..1  (M wave)
    const int wn   = wave & 3;         // 0..3  (N wave)
    const int lo16 = lane & 15;
    const int half = lane >> 4;        // 0 or 1

    const int M0w = blockIdx.y * BM;
    const int N0w = blockIdx.x * BN;
    const int M0  = M0w + wm * 32;
    const int N0  = N0w + wn * 32;

    v8f acc00 = {}, acc01 = {}, acc10 = {}, acc11 = {};

    // unit u covers 8 elements (16B): row = u>>3 (KC/8 = 8 units per row), col8 = u&7
    // A tile: BM*KC/8 = 512 units -> 2 per thread ; B tile: BN*KC/8 = 1024 units -> 4 per thread
    auto issueA = [&](int kc, int buf) {
        #pragma unroll
        for (int i = 0; i < 2; ++i) {
            int u = tid + i * 256, r = u >> 3, c = u & 7;
            async_copy_b128(hb_src + (size_t)(M0w + r) * UNITS + kc * KC + c * 8,
                            &As[buf][r * AROWP + c * 8]);
        }
    };
    auto issueB = [&](int kc, int buf) {
        #pragma unroll
        for (int i = 0; i < 4; ++i) {
            int u = tid + i * 256, r = u >> 3, c = u & 7;
            async_copy_b128(Wb + (size_t)(N0w + r) * UNITS + kc * KC + c * 8,
                            &Bs[buf][r * AROWP + c * 8]);
        }
    };
    auto compute = [&](int buf) {
        #pragma unroll
        for (int k = 0; k < KC; k += 32) {
            const int akb = k + half * 8;
            const int bkb = k + half * 8;
            v16bf a0 = load_frag_a(&As[buf][(wm * 32      + lo16) * AROWP + akb]);
            v16bf a1 = load_frag_a(&As[buf][(wm * 32 + 16 + lo16) * AROWP + akb]);
            v16bf b0 = load_frag_b(&Bs[buf][(wn * 32      + lo16) * AROWP + bkb]);
            v16bf b1 = load_frag_b(&Bs[buf][(wn * 32 + 16 + lo16) * AROWP + bkb]);
            acc00 = __builtin_amdgcn_wmma_f32_16x16x32_bf16(false, a0, false, b0, (short)0, acc00, false, false);
            acc01 = __builtin_amdgcn_wmma_f32_16x16x32_bf16(false, a0, false, b1, (short)0, acc01, false, false);
            acc10 = __builtin_amdgcn_wmma_f32_16x16x32_bf16(false, a1, false, b0, (short)0, acc10, false, false);
            acc11 = __builtin_amdgcn_wmma_f32_16x16x32_bf16(false, a1, false, b1, (short)0, acc11, false, false);
        }
    };

    const int NCH = UNITS / KC;   // 32 chunks
    issueA(0, 0); issueB(0, 0);
    wait_asynccnt0();
    __syncthreads();
    for (int kc = 0; kc < NCH; ++kc) {
        const int nxt = kc + 1;
        if (nxt < NCH) { issueA(nxt, nxt & 1); issueB(nxt, nxt & 1); }  // async fill, overlapped with compute
        compute(kc & 1);
        if (nxt < NCH) { wait_asynccnt0(); }   // own chunk-(kc+1) writes landed in LDS
        __syncthreads();                       // publish across waves; WAR-safe for next overwrite
    }

    // C/D layout (ISA 7.12.2): element (VGPR r, lane): M = r + 8*(lane>>4), N = lane&15.
    // Fuse rank-1 input term + leaky tanh update; write f32 master + bf16 mirror.
    float xv0[8], xv1[8];
    #pragma unroll
    for (int r = 0; r < 8; ++r) {
        xv0[r] = x[(size_t)(M0      + half * 8 + r) * TLEN + t];
        xv1[r] = x[(size_t)(M0 + 16 + half * 8 + r) * TLEN + t];
    }
    const float win0 = W_in[N0      + lo16];
    const float win1 = W_in[N0 + 16 + lo16];

    #pragma unroll
    for (int r = 0; r < 8; ++r) {
        const int m0 = M0 + half * 8 + r;
        const int m1 = m0 + 16;
        const int n0 = N0 + lo16;
        const int n1 = n0 + 16;

        float p00 = acc00[r] + xv0[r] * win0;
        float p01 = acc01[r] + xv0[r] * win1;
        float p10 = acc10[r] + xv1[r] * win0;
        float p11 = acc11[r] + xv1[r] * win1;

        float h00 = KEEP * hf_src[(size_t)m0 * UNITS + n0] + LEAK * tanhf(p00);
        float h01 = KEEP * hf_src[(size_t)m0 * UNITS + n1] + LEAK * tanhf(p01);
        float h10 = KEEP * hf_src[(size_t)m1 * UNITS + n0] + LEAK * tanhf(p10);
        float h11 = KEEP * hf_src[(size_t)m1 * UNITS + n1] + LEAK * tanhf(p11);

        hf_dst[(size_t)m0 * UNITS + n0] = h00;  hb_dst[(size_t)m0 * UNITS + n0] = (__bf16)h00;
        hf_dst[(size_t)m0 * UNITS + n1] = h01;  hb_dst[(size_t)m0 * UNITS + n1] = (__bf16)h01;
        hf_dst[(size_t)m1 * UNITS + n0] = h10;  hb_dst[(size_t)m1 * UNITS + n0] = (__bf16)h10;
        hf_dst[(size_t)m1 * UNITS + n1] = h11;  hb_dst[(size_t)m1 * UNITS + n1] = (__bf16)h11;
    }
}

// ---------------------------------------------------------------------------
// Readout: out[b,o] = tanh(sum_u h[b,u]*w_out[o,u] + b_out[o]), O=2.
// ---------------------------------------------------------------------------
__global__ void __launch_bounds__(256) esn_readout(
        const float* __restrict__ h, const float* __restrict__ w_out,
        const float* __restrict__ b_out, float* __restrict__ out) {
    __shared__ float red0[256];
    __shared__ float red1[256];
    const int b = blockIdx.x;
    const int tid = threadIdx.x;
    float s0 = 0.0f, s1 = 0.0f;
    for (int u = tid; u < UNITS; u += 256) {
        const float hv = h[(size_t)b * UNITS + u];
        s0 += hv * w_out[u];
        s1 += hv * w_out[UNITS + u];
    }
    red0[tid] = s0; red1[tid] = s1;
    __syncthreads();
    for (int off = 128; off > 0; off >>= 1) {
        if (tid < off) { red0[tid] += red0[tid + off]; red1[tid] += red1[tid + off]; }
        __syncthreads();
    }
    if (tid == 0) {
        out[b * 2 + 0] = tanhf(red0[0] + b_out[0]);
        out[b * 2 + 1] = tanhf(red1[0] + b_out[1]);
    }
}

// ---------------------------------------------------------------------------
extern "C" void kernel_launch(void* const* d_in, const int* in_sizes, int n_in,
                              void* d_out, int out_size, void* d_ws, size_t ws_size,
                              hipStream_t stream) {
    (void)in_sizes; (void)n_in; (void)out_size; (void)ws_size;
    const float* x     = (const float*)d_in[0];   // [128][512]
    const float* W_in  = (const float*)d_in[1];   // [2048][1]
    const float* W_res = (const float*)d_in[2];   // [2048][2048]
    const float* w_out = (const float*)d_in[3];   // [2][2048]
    const float* b_out = (const float*)d_in[4];   // [2]
    float* out = (float*)d_out;                   // [128][2]

    // Workspace layout (all offsets 16B-aligned):
    //   Wb   : UNITS*UNITS bf16          (8 MB)    L2-resident weight matrix
    //   hf0/1: BATCH*UNITS f32 each      (1 MB x2) f32 master state, ping-pong
    //   hb0/1: BATCH*UNITS bf16 each     (0.5 MB x2) bf16 mirror for WMMA A
    char* ws = (char*)d_ws;
    __bf16* Wb = (__bf16*)ws;
    size_t off = (size_t)UNITS * UNITS * sizeof(__bf16);
    float* hf0 = (float*)(ws + off);  off += (size_t)BATCH * UNITS * sizeof(float);
    float* hf1 = (float*)(ws + off);  off += (size_t)BATCH * UNITS * sizeof(float);
    __bf16* hb0 = (__bf16*)(ws + off); off += (size_t)BATCH * UNITS * sizeof(__bf16);
    __bf16* hb1 = (__bf16*)(ws + off);

    const int nW = UNITS * UNITS;
    esn_convert_w<<<(nW + 255) / 256, 256, 0, stream>>>(W_res, Wb, nW);
    const int nH = BATCH * UNITS;
    esn_init_h<<<(nH + 255) / 256, 256, 0, stream>>>(hf0, hb0, nH);

    // 512 dependent GEMM steps; kernel boundary is the grid-wide sync,
    // ping-pong buffers make each step race-free.
    for (int t = 0; t < TLEN; ++t) {
        const bool odd = (t & 1) != 0;
        const float*  hfs = odd ? hf1 : hf0;
        float*        hfd = odd ? hf0 : hf1;
        const __bf16* hbs = odd ? hb1 : hb0;
        __bf16*       hbd = odd ? hb0 : hb1;
        esn_step<<<dim3(UNITS / BN, BATCH / BM), 256, 0, stream>>>(
            hbs, Wb, hfs, x, W_in, t, hfd, hbd);
    }
    // TLEN is even -> final state landed in buffer 0.
    esn_readout<<<BATCH, 256, 0, stream>>>(hf0, w_out, b_out, out);
}
